// QuestionEmbeddings_55929064128699
// MI455X (gfx1250) — compile-verified, matrix operand order checked by reading, not run on hardware
//
#include <hip/hip_runtime.h>
#include <hip/hip_bf16.h>

// Problem constants (match reference)
#define SEQ    2048
#define BATCHN 512
#define EMBN   64
#define HIDN   32
#define GATESN 128      // 4*HID
#define NUMQ   1774

typedef __attribute__((ext_vector_type(16))) __bf16 v16bf;
typedef __attribute__((ext_vector_type(2)))  __bf16 v2bf;
typedef __attribute__((ext_vector_type(8)))  float  v8f;

// Workspace layout (bytes)
#define G_OFF   0u              // NUMQ*GATESN floats, permuted gate-column order
#define WT_OFF  (1u << 20)      // 8 tiles * 32 lanes * 8 dwords of packed bf16x2 W_hh B-tiles
#define P_OFF   (2u << 20)      // 32 waves * SEQ * 32 lanes * 2 floats partial batch sums

__device__ __forceinline__ unsigned pack_bf16x2(float a, float b) {
    v2bf v;
    v[0] = (__bf16)a;   // even k -> low 16 bits (each VGPR of the A/B images holds one K pair 2j,2j+1)
    v[1] = (__bf16)b;
    return __builtin_bit_cast(unsigned, v);
}

// Native CDNA5 transcendental tanh. TRANS hazard rule (ISA 7.3): one independent
// op or V_NOP after a TRANS op before its result is used / sources overwritten.
// In a multi-tanh block each v_tanh covers the previous one; one trailing v_nop
// covers the last. Early-clobber outputs so the register allocator cannot alias
// them with the still-live inputs.
__device__ __forceinline__ void tanh2(float& ra, float& rb, float a, float b) {
    asm volatile("v_tanh_f32 %0, %2\n\t"
                 "v_tanh_f32 %1, %3\n\t"
                 "v_nop"
                 : "=&v"(ra), "=&v"(rb)
                 : "v"(a), "v"(b));
}

__device__ __forceinline__ void tanh8(float& r0, float& r1, float& r2, float& r3,
                                      float& r4, float& r5, float& r6, float& r7,
                                      float a0, float a1, float a2, float a3,
                                      float a4, float a5, float a6, float a7) {
    asm volatile("v_tanh_f32 %0, %8\n\t"
                 "v_tanh_f32 %1, %9\n\t"
                 "v_tanh_f32 %2, %10\n\t"
                 "v_tanh_f32 %3, %11\n\t"
                 "v_tanh_f32 %4, %12\n\t"
                 "v_tanh_f32 %5, %13\n\t"
                 "v_tanh_f32 %6, %14\n\t"
                 "v_tanh_f32 %7, %15\n\t"
                 "v_nop"
                 : "=&v"(r0), "=&v"(r1), "=&v"(r2), "=&v"(r3),
                   "=&v"(r4), "=&v"(r5), "=&v"(r6), "=&v"(r7)
                 : "v"(a0), "v"(a1), "v"(a2), "v"(a3),
                   "v"(a4), "v"(a5), "v"(a6), "v"(a7));
}

// ---------------------------------------------------------------------------
// Phase 1a: gate table  G[q][8c+n] = scale*(b_ih[oc]+b_hh[oc]+dot(W_ih[oc,:],emb[q,:]))
// oc = 32*(n%4) + 2c + n/4  (column permutation: each lane's 8 gate values are
// contiguous; each hidden unit's i,f,g,o quadruple is lane-local).
// scale = 0.5 for sigmoid gates (i,f,o): sigma(x) = 0.5 + 0.5*tanh(x/2);
// the 'g' gate (n%4==2) stays unscaled for its direct tanh.
// ---------------------------------------------------------------------------
__global__ void build_gate_table(const float* __restrict__ emb,
                                 const float* __restrict__ W_ih,
                                 const float* __restrict__ b_ih,
                                 const float* __restrict__ b_hh,
                                 float* __restrict__ G) {
    int tid = blockIdx.x * blockDim.x + threadIdx.x;
    if (tid >= NUMQ * GATESN) return;
    int q = tid / GATESN;
    int j = tid % GATESN;                 // stored col: j = 8*c + n
    int c = j >> 3, n = j & 7;
    int gate = n & 3;                     // 0=i 1=f 2=g 3=o
    int oc = 32 * gate + 2 * c + (n >> 2);
    float acc = b_ih[oc] + b_hh[oc];
    const float* w = W_ih + oc * EMBN;
    const float* e = emb + q * EMBN;
#pragma unroll 8
    for (int k = 0; k < EMBN; ++k) acc += w[k] * e[k];
    float scale = (gate == 2) ? 1.0f : 0.5f;
    G[q * GATESN + j] = scale * acc;
}

// ---------------------------------------------------------------------------
// Phase 1b: pre-swizzle W_hh into WMMA B-matrix (32x16 bf16) register images,
// with the same per-gate 0.5 pre-scale as the gate table.
// Layout: Wt[n*256 + lane*8 + v] = pack(bf16 W[oc][klo], bf16 W[oc][klo+1])
// lane<16: v0..3 -> K=0..7, v4..7 -> K=16..23 ; lane>=16: K=8..15, 24..31
// ---------------------------------------------------------------------------
__global__ void build_whh_tiles(const float* __restrict__ W_hh,
                                unsigned* __restrict__ Wt) {
    int tid = blockIdx.x * blockDim.x + threadIdx.x;
    if (tid >= 8 * 32 * 8) return;
    int v    = tid & 7;
    int lane = (tid >> 3) & 31;
    int n    = tid >> 8;
    int c    = lane & 15;
    int gate = n & 3;
    int oc   = 32 * gate + 2 * c + (n >> 2);
    int klo  = (lane < 16) ? ((v < 4) ? 2 * v : 16 + 2 * (v - 4))
                           : ((v < 4) ? 8 + 2 * v : 24 + 2 * (v - 4));
    float scale = (gate == 2) ? 1.0f : 0.5f;
    float a = scale * W_hh[oc * HIDN + klo];
    float b = scale * W_hh[oc * HIDN + klo + 1];
    Wt[tid] = pack_bf16x2(a, b);
}

// ---------------------------------------------------------------------------
// Phase 2: the serial LSTM recurrence. One wave32 per 16-batch tile.
// Per step: gather G rows as the C accumulators, 8x v_wmma_f32_16x16x32_bf16,
// native v_tanh_f32 activations, h -> LDS -> A-layout bf16 reload.
// ---------------------------------------------------------------------------
__global__ void __launch_bounds__(32) lstm_recurrence(
    const int* __restrict__ input,
    const float* __restrict__ G,
    const unsigned* __restrict__ Wt,
    float* __restrict__ P) {

    __shared__ __align__(16) unsigned char hstage[16 * HIDN * 2];  // 1 KB bf16 16x32 row-major

    const int lane = threadIdx.x;        // 0..31
    const int wid  = blockIdx.x;         // 0..31 batch tile
    const int bb   = wid * 16;
    const int cr   = lane & 15;
    const int hi   = lane >> 4;

    union VU { uint4 u[2]; v16bf v; };

    // Resident W_hh B-tiles (8 x 8 VGPRs)
    v16bf Bt[8];
#pragma unroll
    for (int n = 0; n < 8; ++n) {
        VU bu;
        bu.u[0] = *(const uint4*)(Wt + (n * 32 + lane) * 8);
        bu.u[1] = *(const uint4*)(Wt + (n * 32 + lane) * 8 + 4);
        Bt[n] = bu.v;
    }

    // h = 0 (A-matrix image), c = 0
    VU az;
    az.u[0] = make_uint4(0u, 0u, 0u, 0u);
    az.u[1] = make_uint4(0u, 0u, 0u, 0u);
    v16bf A = az.v;

    float cs[16];
#pragma unroll
    for (int i = 0; i < 16; ++i) cs[i] = 0.0f;

    // A-layout LDS read addresses for this lane
    const unsigned a0 = (unsigned)(cr * 64 + (hi ? 16 : 0));
    const unsigned a1 = a0 + 32;

    for (int t = 0; t < SEQ; ++t) {
        // 16 wave-uniform question ids; each half-wave loads its 8 rows
        const int* ip = input + t * BATCHN + bb + hi * 8;
        int4 qa = *(const int4*)(ip);
        int4 qb = *(const int4*)(ip + 4);
        int qq[8] = {qa.x, qa.y, qa.z, qa.w, qb.x, qb.y, qb.z, qb.w};

        // Gather gate-table rows directly into the C accumulators.
        // acc[n] lane layout: (M = r + 8*hi, N = cr) == G[q[M]][8*cr + n]
        v8f acc[8];
#pragma unroll
        for (int r = 0; r < 8; ++r) {
            const float* g = G + (size_t)qq[r] * GATESN + cr * 8;
            float4 ga = *(const float4*)(g);
            float4 gb = *(const float4*)(g + 4);
            acc[0][r] = ga.x; acc[1][r] = ga.y; acc[2][r] = ga.z; acc[3][r] = ga.w;
            acc[4][r] = gb.x; acc[5][r] = gb.y; acc[6][r] = gb.z; acc[7][r] = gb.w;
        }

        // gates += h @ W_hh.T  (8 independent 16x16x32 bf16 WMMAs, in-place C/D)
#pragma unroll
        for (int n = 0; n < 8; ++n) {
            acc[n] = __builtin_amdgcn_wmma_f32_16x16x32_bf16(
                false, A, false, Bt[n], (short)0, acc[n], false, false);
        }

        // Lane-local LSTM cell update via native v_tanh_f32.
        // tile n -> gate n%4 (i,f,g,o), hidden unit k = 2*cr + n/4.
        // i,f,o pre-activations arrive pre-scaled by 0.5: sigma = 0.5 + 0.5*tanh.
        float s0 = 0.0f, s1 = 0.0f;
#pragma unroll
        for (int r = 0; r < 8; ++r) {
            float ti0, ti1, tf0, tf1, tg0, tg1, to0, to1;
            tanh8(ti0, tf0, tg0, to0, ti1, tf1, tg1, to1,
                  acc[0][r], acc[1][r], acc[2][r], acc[3][r],
                  acc[4][r], acc[5][r], acc[6][r], acc[7][r]);

            float i0 = fmaf(0.5f, ti0, 0.5f);
            float f0 = fmaf(0.5f, tf0, 0.5f);
            float o0 = fmaf(0.5f, to0, 0.5f);
            float i1 = fmaf(0.5f, ti1, 0.5f);
            float f1 = fmaf(0.5f, tf1, 0.5f);
            float o1 = fmaf(0.5f, to1, 0.5f);

            float c0 = fmaf(f0, cs[2 * r],     i0 * tg0);
            float c1 = fmaf(f1, cs[2 * r + 1], i1 * tg1);
            cs[2 * r]     = c0;
            cs[2 * r + 1] = c1;

            float tc0, tc1;
            tanh2(tc0, tc1, c0, c1);
            float h0 = o0 * tc0;
            float h1 = o1 * tc1;

            s0 += h0;
            s1 += h1;

            // h row M = r + 8*hi, cols (2cr, 2cr+1) -> bf16 pair in LDS
            *(unsigned*)&hstage[(r + 8 * hi) * 64 + cr * 4] = pack_bf16x2(h0, h1);
        }

        // Partial batch-mean contributions (reduced by finalize kernel);
        // global store overlaps the LDS wait below.
        float2 ps;
        ps.x = s0; ps.y = s1;
        *(float2*)(P + (((size_t)wid * SEQ + t) * 32 + lane) * 2) = ps;

        // Reload h as bf16 A-matrix for the next step (DS ops are wave-in-order)
        asm volatile("s_wait_dscnt 0" ::: "memory");
        VU au;
        au.u[0] = *(const uint4*)&hstage[a0];
        au.u[1] = *(const uint4*)&hstage[a1];
        A = au.v;

        // Prefetch next step's question ids (gfx1250 global_prefetch path)
        int tn = (t + 1 < SEQ) ? (t + 1) : t;
        __builtin_prefetch(input + tn * BATCHN + bb, 0, 0);
    }
}

// ---------------------------------------------------------------------------
// Phase 3: reduce partials -> batch mean -> FC -> tanh (cold, plain math fine)
// pool[k]: k = 2c+slot lives in lanes c and c+16 of every wave
// ---------------------------------------------------------------------------
__global__ void finalize_out(const float* __restrict__ P,
                             const float* __restrict__ fc_w,
                             const float* __restrict__ fc_b,
                             float* __restrict__ out) {
    int t = blockIdx.x * blockDim.x + threadIdx.x;
    if (t >= SEQ) return;
    float pool[HIDN];
#pragma unroll
    for (int k = 0; k < HIDN; ++k) pool[k] = 0.0f;
    for (int w = 0; w < 32; ++w) {
        const float* base = P + (((size_t)w * SEQ + t) * 32) * 2;
#pragma unroll
        for (int c = 0; c < 16; ++c) {
            float2 a = *(const float2*)(base + c * 2);
            float2 b = *(const float2*)(base + (c + 16) * 2);
            pool[2 * c]     += a.x + b.x;
            pool[2 * c + 1] += a.y + b.y;
        }
    }
    float acc = fc_b[0];
#pragma unroll
    for (int k = 0; k < HIDN; ++k) acc += fc_w[k] * (pool[k] * (1.0f / (float)BATCHN));
    out[t] = tanhf(acc);
}

// ---------------------------------------------------------------------------
extern "C" void kernel_launch(void* const* d_in, const int* in_sizes, int n_in,
                              void* d_out, int out_size, void* d_ws, size_t ws_size,
                              hipStream_t stream) {
    const int*   input = (const int*)d_in[0];
    const float* emb   = (const float*)d_in[1];
    const float* W_ih  = (const float*)d_in[2];
    const float* W_hh  = (const float*)d_in[3];
    const float* b_ih  = (const float*)d_in[4];
    const float* b_hh  = (const float*)d_in[5];
    const float* fc_w  = (const float*)d_in[6];
    const float* fc_b  = (const float*)d_in[7];
    float* out = (float*)d_out;

    char* ws = (char*)d_ws;
    float*    G  = (float*)(ws + G_OFF);
    unsigned* Wt = (unsigned*)(ws + WT_OFF);
    float*    P  = (float*)(ws + P_OFF);

    build_gate_table<<<(NUMQ * GATESN + 255) / 256, 256, 0, stream>>>(
        emb, W_ih, b_ih, b_hh, G);
    build_whh_tiles<<<(8 * 32 * 8 + 255) / 256, 256, 0, stream>>>(W_hh, Wt);
    lstm_recurrence<<<BATCHN / 16, 32, 0, stream>>>(input, G, Wt, P);
    finalize_out<<<(SEQ + 127) / 128, 128, 0, stream>>>(P, fc_w, fc_b, out);

    (void)in_sizes; (void)n_in; (void)out_size; (void)ws_size;
}
